// MultiheadAttention_58402965291265
// MI455X (gfx1250) — matrix operand order repaired and checked by reading, not compiled, not run on hardware
//
#include <hip/hip_runtime.h>
#include <hip/hip_bf16.h>
#include <math.h>

// ---------------------------------------------------------------------------
// MHA forward for MI455X (gfx1250, wave32, WMMA).
// x:[2,2048,1024] f32 -> out:[2,2048,1024] f32 ++ attn:[2,16,2048,2048] f32
// ---------------------------------------------------------------------------

#define D_MODEL   1024
#define NUM_HEADS 16
#define DIM_HEAD  64
#define SEQ       2048
#define BATCH     2
#define ROWS_TOT  (BATCH * SEQ)          // 4096
#define BH        (BATCH * NUM_HEADS)    // 32
#define LN_EPS    1e-5f
#define SM_SCALE  0.125f                 // 1/sqrt(64)

typedef _Float16 v16h __attribute__((ext_vector_type(16)));
typedef _Float16 h8   __attribute__((ext_vector_type(8)));
typedef float    v8f  __attribute__((ext_vector_type(8)));

// --- WMMA wrapper: D = A(16x32 f16) * B(32x16 f16) + C(16x16 f32) ----------
__device__ inline v8f wmma32(v16h a, v16h b, v8f c) {
  return __builtin_amdgcn_wmma_f32_16x16x32_f16(
      /*neg_a=*/false, a, /*neg_b=*/false, b,
      /*c_mod=*/(short)0, c, /*reuse_a=*/false, /*reuse_b=*/false);
}

// A-fragment (16x32, MxK), caller passes pointer to this lane's row (m = lane&15).
// ISA layout: lanes kh=(lane>>4): halves 0..7 <- K = k0+8*kh .. +8,
//                                 halves 8..15 <- K = k0+16+8*kh .. +8.
__device__ inline v16h load_a_frag(const _Float16* row, int k0, int kh) {
  h8 lo = *(const h8*)(row + k0 + 8 * kh);
  h8 hi = *(const h8*)(row + k0 + 16 + 8 * kh);
  return __builtin_shufflevector(lo, hi, 0,1,2,3,4,5,6,7,8,9,10,11,12,13,14,15);
}

// B-fragment (32x16, KxN) sourced from N-major storage T[n][k] (ld = K dim).
// Lane holds column n = lane&15, K = k0+8*kh..+8 then k0+16+8*kh..+8 (same as A).
__device__ inline v16h load_b_frag(const _Float16* rowN, int k0, int kh) {
  h8 lo = *(const h8*)(rowN + k0 + 8 * kh);
  h8 hi = *(const h8*)(rowN + k0 + 16 + 8 * kh);
  return __builtin_shufflevector(lo, hi, 0,1,2,3,4,5,6,7,8,9,10,11,12,13,14,15);
}

// ---------------------------------------------------------------------------
// K0: LayerNorm, one block per row of 1024; write xn as f16.
// ---------------------------------------------------------------------------
__global__ __launch_bounds__(256) void k_ln(const float* __restrict__ x,
                                            const float* __restrict__ gamma,
                                            const float* __restrict__ beta,
                                            _Float16* __restrict__ xn) {
  const int row = blockIdx.x;
  const float* xr = x + (size_t)row * D_MODEL;
  const int t = threadIdx.x, lane = t & 31, wave = t >> 5;
  __shared__ float red0[8], red1[8];

  float v[4]; float s = 0.f;
#pragma unroll
  for (int i = 0; i < 4; ++i) { v[i] = xr[t + 256 * i]; s += v[i]; }
#pragma unroll
  for (int off = 16; off; off >>= 1) s += __shfl_xor(s, off, 32);
  if (lane == 0) red0[wave] = s;
  __syncthreads();
  float mu = 0.f;
#pragma unroll
  for (int i = 0; i < 8; ++i) mu += red0[i];
  mu *= (1.0f / D_MODEL);

  float vs = 0.f;
#pragma unroll
  for (int i = 0; i < 4; ++i) { float d = v[i] - mu; vs += d * d; }
#pragma unroll
  for (int off = 16; off; off >>= 1) vs += __shfl_xor(vs, off, 32);
  if (lane == 0) red1[wave] = vs;
  __syncthreads();
  float var = 0.f;
#pragma unroll
  for (int i = 0; i < 8; ++i) var += red1[i];
  const float rstd = rsqrtf(var * (1.0f / D_MODEL) + LN_EPS);

  _Float16* orow = xn + (size_t)row * D_MODEL;
#pragma unroll
  for (int i = 0; i < 4; ++i) {
    int c = t + 256 * i;
    orow[c] = (_Float16)(((v[i] - mu) * rstd) * gamma[c] + beta[c]);
  }
}

// ---------------------------------------------------------------------------
// K1: convert W_qkv [1024,3072] -> Wq_t f16 [3072,1024] (N-major),
//     and  W_out [1024,1024] -> Wo_t f16 [1024,1024] (N-major).
// ---------------------------------------------------------------------------
__global__ __launch_bounds__(256) void k_cvt_w(const float* __restrict__ Wqkv,
                                               const float* __restrict__ Wout,
                                               _Float16* __restrict__ Wqt,
                                               _Float16* __restrict__ Wot) {
  size_t i = (size_t)blockIdx.x * 256 + threadIdx.x;
  const size_t NQ = (size_t)3 * D_MODEL * D_MODEL;          // 3,145,728
  if (i < NQ) {
    int n = (int)(i >> 10), k = (int)(i & 1023);
    Wqt[i] = (_Float16)Wqkv[(size_t)k * (3 * D_MODEL) + n];
  } else {
    size_t j = i - NQ;
    if (j < (size_t)D_MODEL * D_MODEL) {
      int n = (int)(j >> 10), k = (int)(j & 1023);
      Wot[j] = (_Float16)Wout[(size_t)k * D_MODEL + n];
    }
  }
}

// ---------------------------------------------------------------------------
// K2: QKV GEMM [4096,1024]x[1024,3072]; wave = one 16x16 tile, block = 32x64.
// Epilogue scatters into Q [bh][s][64], K [bh][s][64], Vt [bh][64][s] (f16).
// ---------------------------------------------------------------------------
__global__ __launch_bounds__(256) void k_gemm_qkv(const _Float16* __restrict__ A,
                                                  const _Float16* __restrict__ Bt,
                                                  _Float16* __restrict__ Qo,
                                                  _Float16* __restrict__ Ko,
                                                  _Float16* __restrict__ Vto) {
  const int lane = threadIdx.x & 31, wave = threadIdx.x >> 5;
  const int m0 = blockIdx.x * 32 + (wave & 1) * 16;
  const int n0 = blockIdx.y * 64 + (wave >> 1) * 16;
  const int m = lane & 15, kh = lane >> 4;
  const _Float16* arow = A  + (size_t)(m0 + m) * D_MODEL;
  const _Float16* brow = Bt + (size_t)(n0 + m) * D_MODEL;
  v8f c = {};
#pragma unroll 4
  for (int k0 = 0; k0 < D_MODEL; k0 += 32) {
    v16h a = load_a_frag(arow, k0, kh);
    v16h b = load_b_frag(brow, k0, kh);
    c = wmma32(a, b, c);
  }
  const int gc = n0 + m;
  const int sec = gc >> 10, rem = gc & 1023, h = rem >> 6, dh = rem & 63;
#pragma unroll
  for (int r = 0; r < 8; ++r) {
    const int gr = m0 + r + 8 * kh;
    const int bb = gr >> 11, s = gr & 2047;
    const int bh = bb * NUM_HEADS + h;
    const _Float16 v = (_Float16)c[r];
    if (sec == 0)      Qo[((size_t)bh * SEQ + s) * DIM_HEAD + dh] = v;
    else if (sec == 1) Ko[((size_t)bh * SEQ + s) * DIM_HEAD + dh] = v;
    else               Vto[((size_t)bh * DIM_HEAD + dh) * SEQ + s] = v;
  }
}

// ---------------------------------------------------------------------------
// K3: fused attention. Block = 16 query rows x all 2048 keys for one (b,h).
//   phase 1: S = QK^T * scale via WMMA -> LDS f32 [16][2048]  (8 waves)
//   phase 2: softmax per row (wave shfl reductions), write attn f32 to d_out
//            and f16 P to LDS [16][2048]
//   phase 3: O = P @ V via WMMA from LDS + global V^T          (4 waves)
// Dynamic LDS: 16*2048*4 + 16*2048*2 = 192 KB (<= 320 KB / WGP).
// ---------------------------------------------------------------------------
__global__ __launch_bounds__(256) void k_attn(const _Float16* __restrict__ Q,
                                              const _Float16* __restrict__ K,
                                              const _Float16* __restrict__ Vt,
                                              float* __restrict__ attn_out,
                                              _Float16* __restrict__ OH) {
  extern __shared__ char smem[];
  float*    Ss = (float*)smem;                              // [16][2048] f32
  _Float16* Ps = (_Float16*)(smem + 16 * SEQ * sizeof(float)); // [16][2048] f16

  const int bh = blockIdx.y;
  const int s0 = blockIdx.x * 16;
  const int lane = threadIdx.x & 31, wave = threadIdx.x >> 5;
  const int m = lane & 15, kh = lane >> 4;

  const _Float16* Qb = Q  + (size_t)bh * SEQ * DIM_HEAD;
  const _Float16* Kb = K  + (size_t)bh * SEQ * DIM_HEAD;
  const _Float16* Vb = Vt + (size_t)bh * DIM_HEAD * SEQ;

  // ---- phase 1: scores --------------------------------------------------
  const _Float16* qrow = Qb + (size_t)(s0 + m) * DIM_HEAD;
  const v16h a0 = load_a_frag(qrow, 0, kh);
  const v16h a1 = load_a_frag(qrow, 32, kh);
  for (int it = 0; it < SEQ / 128; ++it) {          // 8 waves x 16 cols = 128
    const int n0 = it * 128 + wave * 16;
    const _Float16* krow = Kb + (size_t)(n0 + m) * DIM_HEAD;
    v16h b0 = load_b_frag(krow, 0, kh);
    v16h b1 = load_b_frag(krow, 32, kh);
    v8f c = {};
    c = wmma32(a0, b0, c);
    c = wmma32(a1, b1, c);
    const int col = n0 + m;
#pragma unroll
    for (int r = 0; r < 8; ++r)
      Ss[(size_t)(r + 8 * kh) * SEQ + col] = c[r] * SM_SCALE;
  }
  __syncthreads();

  // ---- phase 2: softmax (2 rows per wave) -------------------------------
#pragma unroll
  for (int rr = 0; rr < 2; ++rr) {
    const int row = wave * 2 + rr;
    float* srow = Ss + (size_t)row * SEQ;
    float mx = -INFINITY;
    for (int c = lane; c < SEQ; c += 32) mx = fmaxf(mx, srow[c]);
#pragma unroll
    for (int off = 16; off; off >>= 1) mx = fmaxf(mx, __shfl_xor(mx, off, 32));
    float sum = 0.f;
    for (int c = lane; c < SEQ; c += 32) {
      float e = __expf(srow[c] - mx);
      srow[c] = e;
      sum += e;
    }
#pragma unroll
    for (int off = 16; off; off >>= 1) sum += __shfl_xor(sum, off, 32);
    const float inv = 1.0f / sum;
    float* grow = attn_out + ((size_t)bh * SEQ + (s0 + row)) * SEQ;
    _Float16* prow = Ps + (size_t)row * SEQ;
    for (int c = lane; c < SEQ; c += 32) {
      float p = srow[c] * inv;
      grow[c] = p;                 // materialized attention weights (output 2)
      prow[c] = (_Float16)p;       // f16 copy for PV WMMA
    }
  }
  __syncthreads();

  // ---- phase 3: O = P @ V (waves 0..3, each one 16-col slice of dh) -----
  if (wave < 4) {
    const int n0 = wave * 16;
    const _Float16* vrow = Vb + (size_t)(n0 + m) * SEQ;   // V^T[n][s]
    const _Float16* prow = Ps + (size_t)m * SEQ;          // P row m
    v8f c = {};
#pragma unroll 4
    for (int k0 = 0; k0 < SEQ; k0 += 32) {
      v16h a = load_a_frag(prow, k0, kh);
      v16h b = load_b_frag(vrow, k0, kh);
      c = wmma32(a, b, c);
    }
    const int bb = bh >> 4, h = bh & 15;
    const int colbase = h * DIM_HEAD + n0 + m;
#pragma unroll
    for (int r = 0; r < 8; ++r) {
      const int row = s0 + r + 8 * kh;
      OH[((size_t)bb * SEQ + row) * D_MODEL + colbase] = (_Float16)c[r];
    }
  }
}

// ---------------------------------------------------------------------------
// K4: output projection [4096,1024]x[1024,1024] -> d_out (f32).
// ---------------------------------------------------------------------------
__global__ __launch_bounds__(256) void k_gemm_out(const _Float16* __restrict__ A,
                                                  const _Float16* __restrict__ Bt,
                                                  float* __restrict__ Out) {
  const int lane = threadIdx.x & 31, wave = threadIdx.x >> 5;
  const int m0 = blockIdx.x * 32 + (wave & 1) * 16;
  const int n0 = blockIdx.y * 64 + (wave >> 1) * 16;
  const int m = lane & 15, kh = lane >> 4;
  const _Float16* arow = A  + (size_t)(m0 + m) * D_MODEL;
  const _Float16* brow = Bt + (size_t)(n0 + m) * D_MODEL;
  v8f c = {};
#pragma unroll 4
  for (int k0 = 0; k0 < D_MODEL; k0 += 32) {
    v16h a = load_a_frag(arow, k0, kh);
    v16h b = load_b_frag(brow, k0, kh);
    c = wmma32(a, b, c);
  }
  const int gc = n0 + m;
#pragma unroll
  for (int r = 0; r < 8; ++r) {
    const int gr = m0 + r + 8 * kh;
    Out[(size_t)gr * D_MODEL + gc] = c[r];
  }
}

// ---------------------------------------------------------------------------
extern "C" void kernel_launch(void* const* d_in, const int* in_sizes, int n_in,
                              void* d_out, int out_size, void* d_ws, size_t ws_size,
                              hipStream_t stream) {
  (void)in_sizes; (void)n_in; (void)out_size; (void)ws_size;
  const float* x     = (const float*)d_in[0];
  const float* gamma = (const float*)d_in[1];
  const float* beta  = (const float*)d_in[2];
  const float* Wqkv  = (const float*)d_in[3];
  const float* Wout  = (const float*)d_in[4];

  float* out0 = (float*)d_out;                             // [2,2048,1024]
  float* attn = out0 + (size_t)BATCH * SEQ * D_MODEL;      // [2,16,2048,2048]

  // workspace carve (f16 buffers), 48 MB total
  char* ws = (char*)d_ws;
  _Float16* xn  = (_Float16*)ws;  ws += (size_t)ROWS_TOT * D_MODEL * 2;       // 8 MB
  _Float16* wqt = (_Float16*)ws;  ws += (size_t)3 * D_MODEL * D_MODEL * 2;    // 6 MB
  _Float16* wot = (_Float16*)ws;  ws += (size_t)D_MODEL * D_MODEL * 2;        // 2 MB
  _Float16* Qb  = (_Float16*)ws;  ws += (size_t)BH * SEQ * DIM_HEAD * 2;      // 8 MB
  _Float16* Kb  = (_Float16*)ws;  ws += (size_t)BH * SEQ * DIM_HEAD * 2;      // 8 MB
  _Float16* Vt  = (_Float16*)ws;  ws += (size_t)BH * SEQ * DIM_HEAD * 2;      // 8 MB
  _Float16* OH  = (_Float16*)ws;  ws += (size_t)ROWS_TOT * D_MODEL * 2;       // 8 MB

  k_ln<<<ROWS_TOT, 256, 0, stream>>>(x, gamma, beta, xn);

  const size_t cvt_total = (size_t)3 * D_MODEL * D_MODEL + (size_t)D_MODEL * D_MODEL;
  k_cvt_w<<<(int)((cvt_total + 255) / 256), 256, 0, stream>>>(Wqkv, Wout, wqt, wot);

  dim3 gq(ROWS_TOT / 32, (3 * D_MODEL) / 64);              // (128, 48)
  k_gemm_qkv<<<gq, 256, 0, stream>>>(xn, wqt, Qb, Kb, Vt);

  const int lds_bytes = 16 * SEQ * (int)sizeof(float) + 16 * SEQ * 2;  // 192 KB
  (void)hipFuncSetAttribute((const void*)k_attn,
                            hipFuncAttributeMaxDynamicSharedMemorySize, lds_bytes);
  dim3 ga(SEQ / 16, BH);                                   // (128, 32)
  k_attn<<<ga, 256, lds_bytes, stream>>>(Qb, Kb, Vt, attn, OH);

  dim3 go(ROWS_TOT / 32, D_MODEL / 64);                    // (128, 16)
  k_gemm_out<<<go, 256, 0, stream>>>(OH, wot, out0);
}
